// MultiHeadedAttention_90374701843255
// MI455X (gfx1250) — matrix operand order repaired and checked by reading, not compiled, not run on hardware
//
#include <hip/hip_runtime.h>
#include <stdint.h>
#include <stddef.h>

// ---------------------------------------------------------------------------
// MultiHeadedAttention (B=1, S=2048, D=1024, H=16, Dk=64) for gfx1250.
// All contractions via v_wmma_f32_16x16x32_bf16 (wave32 WMMA, fp32 accum).
// Attention K/V tiles staged into LDS by the Tensor Data Mover
// (tensor_load_to_lds + s_wait_tensorcnt), double-buffered.
// ---------------------------------------------------------------------------

typedef __attribute__((ext_vector_type(16))) __bf16 v16bf;
typedef __attribute__((ext_vector_type(8)))  __bf16 v8bf;
typedef __attribute__((ext_vector_type(8)))  float  v8f;
typedef __attribute__((ext_vector_type(4)))  unsigned int u32x4;
typedef __attribute__((ext_vector_type(8)))  int    i32x8;
typedef __attribute__((ext_vector_type(4)))  int    i32x4;

#define S_LEN  2048
#define DMODEL 1024
#define NHEAD  16
#define DK     64
#define KB     64                 // keys per attention stage
#define NSTAGE (S_LEN / KB)       // 32

#if defined(__gfx1250__) && __has_builtin(__builtin_amdgcn_tensor_load_to_lds) && \
    __has_builtin(__builtin_amdgcn_s_wait_tensorcnt)
#define USE_TDM 1
#else
#define USE_TDM 0
#endif

// ===========================================================================
// Kernel 1: projections.  grid.x = 4 mats * 16 heads * 2 dk-halves = 128.
// Stage W slice into LDS pre-swizzled into WMMA B-fragment order; 8 waves
// sweep 16-row tiles of X with WMMA.
//   mat 0 -> wsQ (f32, [h][s][64])     mat 1 -> wsK (f32)
//   mat 2 -> vT  (bf16, [h][64][s])    mat 3 -> wsB (f32, b_rotate)
// ===========================================================================
__global__ __launch_bounds__(256) void proj_kernel(
    const float* __restrict__ q_in, const float* __restrict__ k_in,
    const float* __restrict__ v_in, const float* __restrict__ b_in,
    const float* __restrict__ W3,   const float* __restrict__ bproj,
    float* __restrict__ wsQ, float* __restrict__ wsK, float* __restrict__ wsB,
    __bf16* __restrict__ vT)
{
  __shared__ __bf16 Wlds[32 * 2 * 32 * 16];   // 64 KB

  const int mat = blockIdx.x & 3;
  const int h   = (blockIdx.x >> 2) & 15;
  const int ch  = blockIdx.x >> 6;
  const int colbase = 32 * ch;

  const int tid  = threadIdx.x;
  const int lane = tid & 31, wave = tid >> 5;
  const int half = lane >> 4, ln16 = lane & 15;

  const float* X = (mat == 0) ? q_in : (mat == 1) ? k_in : (mat == 2) ? v_in : b_in;
  const float* W = (mat == 3) ? bproj : (W3 + (size_t)mat * DMODEL * NHEAD * DK);

  for (int e = tid; e < 32 * 2 * 32 * 16; e += 256) {
    const int j = e & 15;
    const int L = (e >> 4) & 31;
    const int t = (e >> 9) & 1;
    const int c = e >> 10;
    const int d   = 32 * c + 8 * (L >> 4) + (j & 7) + 16 * (j >> 3);
    const int col = colbase + 16 * t + (L & 15);
    Wlds[e] = (__bf16)W[((size_t)d * NHEAD + h) * DK + col];
  }
  __syncthreads();

  for (int rt = wave; rt < S_LEN / 16; rt += 8) {
    const int rbase = rt * 16;
    v8f acc[2] = {};
    const float* xr = X + (size_t)(rbase + ln16) * DMODEL;

    for (int c = 0; c < 32; ++c) {
      const int klo = 32 * c + (half ? 8 : 0);
      v16bf a;
      #pragma unroll
      for (int i = 0; i < 8; ++i) {
        a[i]     = (__bf16)xr[klo + i];
        a[i + 8] = (__bf16)xr[klo + 16 + i];
      }
      #pragma unroll
      for (int t = 0; t < 2; ++t) {
        const v16bf b = *(const v16bf*)&Wlds[((((c << 1) | t) * 32) + lane) << 4];
        acc[t] = __builtin_amdgcn_wmma_f32_16x16x32_bf16(
                     false, a, false, b, (short)0, acc[t], false, false);
      }
    }

    if (mat == 2) {
      #pragma unroll
      for (int t = 0; t < 2; ++t) {
        v8bf p;
        #pragma unroll
        for (int r = 0; r < 8; ++r) p[r] = (__bf16)acc[t][r];
        const int col = colbase + 16 * t + ln16;
        *(v8bf*)&vT[((size_t)h * DK + col) * S_LEN + rbase + 8 * half] = p;
      }
    } else {
      float* dst = (mat == 0) ? wsQ : (mat == 1) ? wsK : wsB;
      #pragma unroll
      for (int t = 0; t < 2; ++t)
        #pragma unroll
        for (int r = 0; r < 8; ++r)
          dst[((size_t)h * S_LEN + rbase + r + 8 * half) * DK + colbase + 16 * t + ln16]
              = acc[t][r];
    }
  }
}

// ===========================================================================
// Kernel 2: RoPE * b_rotate on q and k, emit bf16 qh/kh ([h][s][64]).
// ===========================================================================
__global__ __launch_bounds__(256) void rope_kernel(
    const float* __restrict__ wsQ, const float* __restrict__ wsK,
    const float* __restrict__ wsB,
    __bf16* __restrict__ qh, __bf16* __restrict__ kh)
{
  const int idx = blockIdx.x * 256 + threadIdx.x;
  if (idx >= NHEAD * S_LEN * 32) return;
  const int i = idx & 31;
  const int s = (idx >> 5) & (S_LEN - 1);
  const int h = idx >> 16;

  const size_t base = ((size_t)h * S_LEN + s) * DK + 2 * i;
  const float inv = expf(-0.28782313662425572f * (float)i);   // 10000^(-i/32)
  const float ang = (float)s * inv;
  float sn, cs;
  sincosf(ang, &sn, &cs);

  const float be = wsB[base], bo = wsB[base + 1];
  const float q1 = wsQ[base], q2 = wsQ[base + 1];
  qh[base]     = (__bf16)((q1 * cs - q2 * sn) * be);
  qh[base + 1] = (__bf16)((q1 * sn + q2 * cs) * bo);
  const float k1 = wsK[base], k2 = wsK[base + 1];
  kh[base]     = (__bf16)((k1 * cs - k2 * sn) * be);
  kh[base + 1] = (__bf16)((k1 * sn + k2 * cs) * bo);
}

// ===========================================================================
// TDM 2D tile load: build D# per CDNA5 ISA §8 and issue tensor_load_to_lds
// (6-arg toolchain form: g0, g1, g2, g3, extra group, cpol).
//   width  = tensor_dim0 = tile_dim0 (elements, 2B each)
//   rows   = tensor_dim1 = tile_dim1
//   stride = tensor_dim0_stride (elements)
// Generic LDS pointers carry the LDS byte offset in their low 32 bits
// (flat-aperture mapping).
// ===========================================================================
#if USE_TDM
static __device__ inline void tdm_load_2d(const void* gptr, uint32_t lds_off,
                                          uint32_t width, uint32_t rows,
                                          uint32_t stride)
{
  const uint64_t ga = (uint64_t)(uintptr_t)gptr;
  u32x4 g0;
  g0[0] = 1u;                                               // count=1 (valid)
  g0[1] = lds_off;                                          // lds_addr
  g0[2] = (uint32_t)ga;                                     // global_addr lo
  g0[3] = ((uint32_t)(ga >> 32) & 0x01FFFFFFu) | 0x80000000u; // ga hi | type=2
  i32x8 g1;
  g1[0] = (int)(1u << 16);                                  // data_size = 2B
  g1[1] = (int)((width & 0xFFFFu) << 16);                   // tensor_dim0 lo
  g1[2] = (int)((width >> 16) | ((rows & 0xFFFFu) << 16));  // dim0 hi | dim1 lo
  g1[3] = (int)((rows >> 16) | ((width & 0xFFFFu) << 16));  // dim1 hi | tile_dim0
  g1[4] = (int)(rows & 0xFFFFu);                            // tile_dim1 (dim2=0)
  g1[5] = (int)stride;                                      // dim0_stride lo
  g1[6] = 0;                                                // dim0_stride hi
  g1[7] = 0;
  const i32x4 z4 = {0, 0, 0, 0};
  const i32x8 z8 = {0, 0, 0, 0, 0, 0, 0, 0};
  __builtin_amdgcn_tensor_load_to_lds(g0, g1, z4, z4, z8, 0);
}
#endif

// Fallback: cooperative copy of one 64x64 bf16 K tile + one 64x64 V tile.
static __device__ inline void coop_stage(const __bf16* __restrict__ krow0,
                                         const __bf16* __restrict__ vrow0,
                                         __bf16* kdst, __bf16* vdst,
                                         int kv, int tid)
{
  for (int idx = tid; idx < 512; idx += 256) {
    const int row = idx >> 3, c = (idx & 7) * 8;
    *(v8bf*)&kdst[row * 64 + c] = *(const v8bf*)&krow0[(size_t)(kv + row) * DK + c];
    *(v8bf*)&vdst[row * 64 + c] = *(const v8bf*)&vrow0[(size_t)row * S_LEN + kv + c];
  }
}

// ===========================================================================
// Kernel 3: flash attention.  grid = (S/128, H), 256 threads = 8 waves,
// one 16-query tile per wave.  K/V tiles (64 keys) staged in LDS via TDM,
// double-buffered; per stage 8 score WMMAs + online softmax + 8 PV WMMAs.
// ===========================================================================
__global__ __launch_bounds__(256) void attn_kernel(
    const __bf16* __restrict__ qh, const __bf16* __restrict__ kh,
    const __bf16* __restrict__ vT, const uint8_t* __restrict__ maskp,
    float* __restrict__ out)
{
  __shared__ __bf16 Kt[2][KB * DK];       // 2 x 8 KB, row-major [key][dk]
  __shared__ __bf16 Vt[2][DK * KB];       // 2 x 8 KB, row-major [dk][key]
  __shared__ __bf16 Pbuf[8][16 * KB];     // 16 KB, per-wave P tile

  const int tid  = threadIdx.x;
  const int lane = tid & 31, wave = tid >> 5;
  const int half = lane >> 4, ln16 = lane & 15;
  const int h     = blockIdx.y;
  const int qbase = blockIdx.x * 128 + wave * 16;

  const __bf16* krow0 = kh + (size_t)h * S_LEN * DK;
  const __bf16* vrow0 = vT + (size_t)h * DK * S_LEN;

  // Preload q A-fragments (two 32-wide dk chunks).
  v16bf qf[2];
  {
    const __bf16* qr = qh + ((size_t)h * S_LEN + qbase + ln16) * DK;
    #pragma unroll
    for (int dc = 0; dc < 2; ++dc) {
      const int klo = dc * 32 + (half ? 16 : 0);
      const v8bf lo = *(const v8bf*)(qr + klo);
      const v8bf hi = *(const v8bf*)(qr + klo + 8);
      qf[dc] = __builtin_shufflevector(lo, hi, 0,1,2,3,4,5,6,7,8,9,10,11,12,13,14,15);
    }
  }

  float qm[8];
  #pragma unroll
  for (int r = 0; r < 8; ++r)
    qm[r] = maskp[qbase + r + 8 * half] ? 1.f : 0.f;

  float m[8], l[8];
  v8f O[4] = {};
  #pragma unroll
  for (int r = 0; r < 8; ++r) { m[r] = -3.0e38f; l[r] = 0.f; }

  // Prologue: stage 0 into buffer 0.
#if USE_TDM
  if (wave == 0) {
    tdm_load_2d(krow0, (uint32_t)(uintptr_t)&Kt[0][0], DK, KB, DK);
    tdm_load_2d(vrow0, (uint32_t)(uintptr_t)&Vt[0][0], KB, DK, S_LEN);
  }
#else
  coop_stage(krow0, vrow0, &Kt[0][0], &Vt[0][0], 0, tid);
#endif

  for (int si = 0; si < NSTAGE; ++si) {
    const int buf = si & 1;
    const int kv  = si * KB;

    // Kick off DMA for the next stage into the other buffer, then wait for
    // this stage's two TDM ops (in-order per wave) and rendezvous.
#if USE_TDM
    if (wave == 0) {
      if (si + 1 < NSTAGE) {
        tdm_load_2d(krow0 + (size_t)(kv + KB) * DK,
                    (uint32_t)(uintptr_t)&Kt[buf ^ 1][0], DK, KB, DK);
        tdm_load_2d(vrow0 + (kv + KB),
                    (uint32_t)(uintptr_t)&Vt[buf ^ 1][0], KB, DK, S_LEN);
        __builtin_amdgcn_s_wait_tensorcnt(2);
      } else {
        __builtin_amdgcn_s_wait_tensorcnt(0);
      }
    }
#else
    if (si + 1 < NSTAGE)
      coop_stage(krow0 + (size_t)(kv + KB) * DK, vrow0 + (kv + KB),
                 &Kt[buf ^ 1][0], &Vt[buf ^ 1][0], 0, tid);
#endif
    __syncthreads();

    // ---- scores for 4 sub-tiles of 16 keys ----
    float sv[4][8];
    #pragma unroll
    for (int st = 0; st < 4; ++st) {
      const int klocal = st * 16 + ln16;                  // B: lane -> key row N
      const float kmv = maskp[kv + klocal] ? 1.f : 0.f;
      const __bf16* kp = &Kt[buf][klocal * DK + (half ? 16 : 0)];
      const v16bf b0 = *(const v16bf*)(kp);               // dk 0..31
      const v16bf b1 = *(const v16bf*)(kp + 32);          // dk 32..63
      v8f c = {};
      c = __builtin_amdgcn_wmma_f32_16x16x32_bf16(false, qf[0], false, b0, (short)0, c, false, false);
      c = __builtin_amdgcn_wmma_f32_16x16x32_bf16(false, qf[1], false, b1, (short)0, c, false, false);
      #pragma unroll
      for (int r = 0; r < 8; ++r)
        sv[st][r] = (qm[r] * kmv != 0.f) ? c[r] * 0.125f : -1.0e30f;
    }

    // ---- online softmax (row r+8*half lives in the 16 lanes of this half) ----
    float mnew[8], alpha[8];
    #pragma unroll
    for (int r = 0; r < 8; ++r) {
      float t = fmaxf(fmaxf(sv[0][r], sv[1][r]), fmaxf(sv[2][r], sv[3][r]));
      #pragma unroll
      for (int off = 1; off <= 8; off <<= 1)
        t = fmaxf(t, __shfl_xor(t, off, 32));
      mnew[r]  = fmaxf(m[r], t);
      alpha[r] = expf(m[r] - mnew[r]);
      m[r]     = mnew[r];
    }

    float p[4][8];
    #pragma unroll
    for (int st = 0; st < 4; ++st)
      #pragma unroll
      for (int r = 0; r < 8; ++r)
        p[st][r] = expf(sv[st][r] - mnew[r]);

    #pragma unroll
    for (int r = 0; r < 8; ++r) {
      float t = (p[0][r] + p[1][r]) + (p[2][r] + p[3][r]);
      #pragma unroll
      for (int off = 1; off <= 8; off <<= 1)
        t += __shfl_xor(t, off, 32);
      l[r] = l[r] * alpha[r] + t;
    }

    #pragma unroll
    for (int t4 = 0; t4 < 4; ++t4)
      #pragma unroll
      for (int r = 0; r < 8; ++r)
        O[t4][r] *= alpha[r];

    // ---- P: C layout -> per-wave LDS tile [16][64] -> A fragments ----
    #pragma unroll
    for (int st = 0; st < 4; ++st)
      #pragma unroll
      for (int r = 0; r < 8; ++r)
        Pbuf[wave][(r + 8 * half) * KB + st * 16 + ln16] = (__bf16)p[st][r];

    #pragma unroll
    for (int kc = 0; kc < 2; ++kc) {
      const int k0 = kc * 32 + (half ? 16 : 0);
      const v8bf lo = *(const v8bf*)&Pbuf[wave][ln16 * KB + k0];
      const v8bf hi = *(const v8bf*)&Pbuf[wave][ln16 * KB + k0 + 8];
      const v16bf pf = __builtin_shufflevector(lo, hi,
          0,1,2,3,4,5,6,7,8,9,10,11,12,13,14,15);
      #pragma unroll
      for (int t4 = 0; t4 < 4; ++t4) {
        const int col = t4 * 16 + ln16;
        const v16bf vf = *(const v16bf*)&Vt[buf][col * KB + kc * 32 + (half ? 16 : 0)];
        O[t4] = __builtin_amdgcn_wmma_f32_16x16x32_bf16(
                    false, pf, false, vf, (short)0, O[t4], false, false);
      }
    }

    __syncthreads();   // all waves done with buf before it is re-filled
  }

  // Epilogue: normalize, write out [s][h*64 + col].
  #pragma unroll
  for (int t4 = 0; t4 < 4; ++t4)
    #pragma unroll
    for (int r = 0; r < 8; ++r)
      out[(size_t)(qbase + r + 8 * half) * (NHEAD * DK) + h * DK + t4 * 16 + ln16]
          = O[t4][r] / l[r];
}

// ===========================================================================
// Launch.  ws layout (bytes):
//   [0,8M)   wsQ f32   [8M,16M) wsK f32   [16M,24M) wsB f32
//   [24M,28M) qh bf16  [28M,32M) kh bf16  [32M,36M) vT bf16
// ===========================================================================
extern "C" void kernel_launch(void* const* d_in, const int* in_sizes, int n_in,
                              void* d_out, int out_size, void* d_ws, size_t ws_size,
                              hipStream_t stream)
{
  const float*   q_in  = (const float*)d_in[0];
  const float*   k_in  = (const float*)d_in[1];
  const float*   v_in  = (const float*)d_in[2];
  const float*   b_in  = (const float*)d_in[3];
  const uint8_t* maskp = (const uint8_t*)d_in[4];   // bool mask, 1 byte/elem
  const float*   W3    = (const float*)d_in[5];     // (3, 1024, 16, 64)
  const float*   bproj = (const float*)d_in[6];     // (1024, 16, 64)
  float*         outp  = (float*)d_out;

  char* ws = (char*)d_ws;
  const size_t MB = 1ull << 20;
  float*  wsQ = (float*)(ws + 0 * MB);
  float*  wsK = (float*)(ws + 8 * MB);
  float*  wsB = (float*)(ws + 16 * MB);
  __bf16* qh  = (__bf16*)(ws + 24 * MB);
  __bf16* kh  = (__bf16*)(ws + 28 * MB);
  __bf16* vT  = (__bf16*)(ws + 32 * MB);

  proj_kernel<<<dim3(128), dim3(256), 0, stream>>>(
      q_in, k_in, v_in, b_in, W3, bproj, wsQ, wsK, wsB, vT);

  rope_kernel<<<dim3((NHEAD * S_LEN * 32) / 256), dim3(256), 0, stream>>>(
      wsQ, wsK, wsB, qh, kh);

  attn_kernel<<<dim3(S_LEN / 128, NHEAD), dim3(256), 0, stream>>>(
      qh, kh, vT, maskp, outp);
}
